// SceneCaptionModule_53558242181152
// MI455X (gfx1250) — matrix operand order repaired and checked by reading, not compiled, not run on hardware
//
#include <hip/hip_runtime.h>
#include <hip/hip_bf16.h>
#include <math.h>

// Problem dims
#define B_    256
#define T_    32
#define P_    256
#define EMB_  300
#define FEAT_ 128
#define VOC_  3433

#define KCH   10          // K chunks of 32 covering 320 (pad of 300)
#define KCHM  4           // K chunks for FEAT=128
#define JT    19          // EMB tiles of 16 (304)
#define VT    215         // vocab tiles of 16 (3440)
#define HS    304         // h fp32 row stride (floats)
#define MT    16          // batch tiles (256/16)

typedef __attribute__((ext_vector_type(16))) __bf16        v16bf;
typedef __attribute__((ext_vector_type(8)))  float         v8f;
typedef __attribute__((ext_vector_type(8)))  unsigned int  v8u;

// ---------- helpers ----------
__device__ __forceinline__ unsigned short f2bf(float f) {
    unsigned u = __float_as_uint(f);
    unsigned r = u + 0x7FFFu + ((u >> 16) & 1u);   // RNE to bf16
    return (unsigned short)(r >> 16);
}
__device__ __forceinline__ unsigned pk2(float a, float b) {
    return (unsigned)f2bf(a) | ((unsigned)f2bf(b) << 16);
}
__device__ __forceinline__ float sigm(float x) { return 1.0f / (1.0f + __expf(-x)); }

// Fragment = 256 dwords; dword index = lane*8 + v  (lane-contiguous -> 2x b128/lane)
__device__ __forceinline__ v8u load_frag(const unsigned* __restrict__ p, int lane) {
    const uint4* q = (const uint4*)(p + lane * 8);
    uint4 lo = q[0], hi = q[1];
    v8u r = {lo.x, lo.y, lo.z, lo.w, hi.x, hi.y, hi.z, hi.w};
    return r;
}
__device__ __forceinline__ void store_frag(unsigned* __restrict__ p, int lane, v8u d) {
    uint4* q = (uint4*)(p + lane * 8);
    q[0] = make_uint4(d[0], d[1], d[2], d[3]);
    q[1] = make_uint4(d[4], d[5], d[6], d[7]);
}

__device__ __forceinline__ v8f wmma_bf16(v8u a, v8u b, v8f c) {
    return __builtin_amdgcn_wmma_f32_16x16x32_bf16(
        false, __builtin_bit_cast(v16bf, a),
        false, __builtin_bit_cast(v16bf, b),
        (short)0, c, false, false);
}

// A-matrix 16x32 bf16 per-VGPR K offset (ISA 7.12.2): lane l -> M=l&15, hi=l>>4,
// VGPR v holds K = (v<4?0:16) + hi*8 + (v&3)*2 and +1 within the 32-chunk.
__device__ __forceinline__ int a_kbase(int v, int hi) {
    return ((v < 4) ? 0 : 16) + hi * 8 + (v & 3) * 2;
}

// ---------- weight packing (once per launch) ----------
// Gate weights: B = W^T, W is [3*EMB, EMB]; gate-split so N tiles stay 16-aligned.
// frag f = ((g*JT + jt)*KCH + kc);  dword = f*256 + lane*8 + v
// B element (k, n): lane = hi*16 + n%16; vgpr v: k = kc*32 + a_kbase(v, hi) (same K order as A)
__global__ void pack_gates_kernel(const float* __restrict__ W, unsigned* __restrict__ dst) {
    int id = blockIdx.x * blockDim.x + threadIdx.x;
    const int total = 3 * JT * KCH * 256;
    if (id >= total) return;
    int d = id & 255, f = id >> 8;
    int lane = d >> 3, v = d & 7;
    int kc = f % KCH, jt = (f / KCH) % JT, g = f / (KCH * JT);
    int n = jt * 16 + (lane & 15);
    int k = kc * 32 + a_kbase(v, lane >> 4);
    float a = (n < EMB_ && k     < EMB_) ? W[(g * EMB_ + n) * EMB_ + k]     : 0.0f;
    float b = (n < EMB_ && k + 1 < EMB_) ? W[(g * EMB_ + n) * EMB_ + k + 1] : 0.0f;
    dst[f * 256 + d] = pk2(a, b);
}

// Direct (non-transposed) pack for [K, N] matrices (W_cls, W_map). frag f = nt*kChunks + kc
__global__ void pack_direct_kernel(const float* __restrict__ W, unsigned* __restrict__ dst,
                                   int K, int N, int nTiles, int kChunks) {
    int id = blockIdx.x * blockDim.x + threadIdx.x;
    int total = nTiles * kChunks * 256;
    if (id >= total) return;
    int d = id & 255, f = id >> 8;
    int lane = d >> 3, v = d & 7;
    int kc = f % kChunks, nt = f / kChunks;
    int n = nt * 16 + (lane & 15);
    int k = kc * 32 + a_kbase(v, lane >> 4);
    float a = (k     < K && n < N) ? W[k * N + n]       : 0.0f;
    float b = (k + 1 < K && n < N) ? W[(k + 1) * N + n] : 0.0f;
    dst[f * 256 + d] = pk2(a, b);
}

// Pack lang_feat (teacher-forced steps 0..T-2) into A fragments.
// frag f = ((t*MT + mt)*KCH + kc)
__global__ void pack_x_kernel(const float* __restrict__ lang, unsigned* __restrict__ Xfrag) {
    int id = blockIdx.x * blockDim.x + threadIdx.x;
    const int total = (T_ - 1) * MT * KCH * 256;
    if (id >= total) return;
    int d = id & 255, f = id >> 8;
    int lane = d >> 3, v = d & 7;
    int kc = f % KCH, mt = (f / KCH) % MT, t = f / (KCH * MT);
    int m = mt * 16 + (lane & 15);
    int k = kc * 32 + a_kbase(v, lane >> 4);
    const float* row = lang + ((size_t)m * T_ + t) * EMB_;
    float a = (k     < EMB_) ? row[k]     : 0.0f;
    float b = (k + 1 < EMB_) ? row[k + 1] : 0.0f;
    Xfrag[(size_t)f * 256 + d] = pk2(a, b);
}

// Repack h fp32 [256,HS] into A fragments (used once, after h0). frag f = mt*KCH + kc
__global__ __launch_bounds__(32)
void repack_h_kernel(const float* __restrict__ H, unsigned* __restrict__ Hfrag) {
    int kc = blockIdx.x, mt = blockIdx.y, lane = threadIdx.x;
    int m = mt * 16 + (lane & 15), hi = lane >> 4;
    const float* row = H + (size_t)m * HS;
    v8u d;
#pragma unroll
    for (int v = 0; v < 8; ++v) {
        int j = kc * 32 + a_kbase(v, hi);
        float a = (j     < EMB_) ? row[j]     : 0.0f;
        float b = (j + 1 < EMB_) ? row[j + 1] : 0.0f;
        d[v] = pk2(a, b);
    }
    store_frag(Hfrag + (size_t)(mt * KCH + kc) * 256, lane, d);
}

// ---------- h0 = relu(gather(obj) @ W_map + b_map) ----------
__global__ __launch_bounds__(32)
void h0_kernel(const float* __restrict__ obj, const int* __restrict__ tgt,
               const unsigned* __restrict__ WmapP, const float* __restrict__ bmap,
               float* __restrict__ H) {
    int jt = blockIdx.x, mt = blockIdx.y, lane = threadIdx.x;
    int m0 = mt * 16, j0 = jt * 16;
    v8f acc = {0.f, 0.f, 0.f, 0.f, 0.f, 0.f, 0.f, 0.f};
    int mrow = m0 + (lane & 15), hi = lane >> 4;
    const float* arow = obj + ((size_t)mrow * P_ + tgt[mrow]) * FEAT_;
#pragma unroll
    for (int kc = 0; kc < KCHM; ++kc) {
        v8u a;   // on-the-fly conversion (runs once, K=128 always in range)
#pragma unroll
        for (int v = 0; v < 8; ++v) {
            float2 f = *(const float2*)(arow + kc * 32 + a_kbase(v, hi));
            a[v] = pk2(f.x, f.y);
        }
        v8u b = load_frag(WmapP + (jt * KCHM + kc) * 256, lane);
        acc = wmma_bf16(a, b, acc);
    }
    int nl = lane & 15, hb = (lane >> 4) * 8;
#pragma unroll
    for (int r = 0; r < 8; ++r) {
        int m = m0 + r + hb, j = j0 + nl;
        float x = (j < EMB_) ? fmaxf(acc[r] + bmap[j], 0.0f) : 0.0f;
        H[m * HS + j] = x;
    }
}

// ---------- fused GRU step ----------
// Each wave owns a 16(batch) x 32(EMB) output region = one full A fragment of h_new:
// 12 WMMA accumulators (2 j-tiles x {i_r,i_z,i_n,h_r,h_z,h_n}), gate math, then an
// LDS cross-lane transpose to emit h_new both as fp32 rows and as a packed bf16 fragment.
__global__ __launch_bounds__(32)
void gru_step_kernel(const unsigned* __restrict__ Xfrag, int t,
                     const unsigned* __restrict__ WihP, const unsigned* __restrict__ WhhP,
                     const float* __restrict__ bih, const float* __restrict__ bhh,
                     const float* __restrict__ Hcur, const unsigned* __restrict__ HfragCur,
                     float* __restrict__ Hnxt, unsigned* __restrict__ HfragNxt) {
    __shared__ float tileLds[16 * 32];
    int kco = blockIdx.x;            // output K-chunk: j in [kco*32, kco*32+32)
    int mt  = blockIdx.y;
    int lane = threadIdx.x;
    int jt0 = 2 * kco;
    bool has1 = (jt0 + 1) < JT;      // tile 19 doesn't exist (JT=19 odd)
    v8f z8 = {0.f, 0.f, 0.f, 0.f, 0.f, 0.f, 0.f, 0.f};
    v8f air[2] = {z8, z8}, aiz[2] = {z8, z8}, ain[2] = {z8, z8};
    v8f ahr[2] = {z8, z8}, ahz[2] = {z8, z8}, ahn[2] = {z8, z8};

    const unsigned* xb = Xfrag   + (size_t)((t * MT + mt) * KCH) * 256;
    const unsigned* hf = HfragCur + (size_t)(mt * KCH) * 256;
#pragma unroll 2
    for (int kc = 0; kc < KCH; ++kc) {
        v8u ax = load_frag(xb + kc * 256, lane);
        v8u ah = load_frag(hf + kc * 256, lane);
#pragma unroll
        for (int s = 0; s < 2; ++s) {
            if (s == 0 || has1) {
                int jt = jt0 + s;
                air[s] = wmma_bf16(ax, load_frag(WihP + ((0 * JT + jt) * KCH + kc) * 256, lane), air[s]);
                aiz[s] = wmma_bf16(ax, load_frag(WihP + ((1 * JT + jt) * KCH + kc) * 256, lane), aiz[s]);
                ain[s] = wmma_bf16(ax, load_frag(WihP + ((2 * JT + jt) * KCH + kc) * 256, lane), ain[s]);
                ahr[s] = wmma_bf16(ah, load_frag(WhhP + ((0 * JT + jt) * KCH + kc) * 256, lane), ahr[s]);
                ahz[s] = wmma_bf16(ah, load_frag(WhhP + ((1 * JT + jt) * KCH + kc) * 256, lane), ahz[s]);
                ahn[s] = wmma_bf16(ah, load_frag(WhhP + ((2 * JT + jt) * KCH + kc) * 256, lane), ahn[s]);
            }
        }
    }

    // Gate math: C layout -> lane l, reg r holds (m = r + 8*(l>>4), n = l&15)
    int nl = lane & 15, hb = (lane >> 4) * 8;
    int m0 = mt * 16;
#pragma unroll
    for (int s = 0; s < 2; ++s) {
        int j0 = (jt0 + s) * 16;
#pragma unroll
        for (int r = 0; r < 8; ++r) {
            int m = m0 + r + hb, j = j0 + nl;
            float out = 0.0f;
            if ((s == 0 || has1) && j < EMB_) {
                float rr = sigm(air[s][r] + ahr[s][r] + bih[j] + bhh[j]);
                float zz = sigm(aiz[s][r] + ahz[s][r] + bih[EMB_ + j] + bhh[EMB_ + j]);
                float nn = tanhf(ain[s][r] + bih[2 * EMB_ + j] + rr * (ahn[s][r] + bhh[2 * EMB_ + j]));
                float ho = Hcur[m * HS + j];
                out = (1.0f - zz) * nn + zz * ho;
            }
            if (j < HS) Hnxt[m * HS + j] = out;
            tileLds[(r + hb) * 32 + (s * 16 + nl)] = out;
        }
    }
    __syncthreads();   // single wave: LDS is in-order; keeps compiler honest

    // Emit the 16x32 tile as one packed bf16 A-fragment (frag = mt*KCH + kco)
    int mfr = lane & 15, hi = lane >> 4;
    v8u d;
#pragma unroll
    for (int v = 0; v < 8; ++v) {
        int base = a_kbase(v, hi);
        d[v] = pk2(tileLds[mfr * 32 + base], tileLds[mfr * 32 + base + 1]);
    }
    store_frag(HfragNxt + (size_t)(mt * KCH + kco) * 256, lane, d);
}

// ---------- logits = h @ W_cls + b_cls, 2 N-tiles per wave, zero-conversion A ----------
__global__ __launch_bounds__(32)
void logits_kernel(const unsigned* __restrict__ Hfrag, const unsigned* __restrict__ WclsP,
                   const float* __restrict__ bcls, float* __restrict__ out, int t) {
    int ng = blockIdx.x, mt = blockIdx.y, lane = threadIdx.x;
    int m0 = mt * 16;
    int nt0 = ng * 2;
    bool has1 = (nt0 + 1) < VT;
    v8f acc0 = {0.f, 0.f, 0.f, 0.f, 0.f, 0.f, 0.f, 0.f};
    v8f acc1 = acc0;
    const unsigned* hf = Hfrag + (size_t)(mt * KCH) * 256;
#pragma unroll 2
    for (int kc = 0; kc < KCH; ++kc) {
        v8u a = load_frag(hf + kc * 256, lane);
        acc0 = wmma_bf16(a, load_frag(WclsP + ((nt0 + 0) * KCH + kc) * 256, lane), acc0);
        if (has1)
            acc1 = wmma_bf16(a, load_frag(WclsP + ((nt0 + 1) * KCH + kc) * 256, lane), acc1);
    }
    int nl = lane & 15, hb = (lane >> 4) * 8;
#pragma unroll
    for (int r = 0; r < 8; ++r) {
        int b = m0 + r + hb;
        size_t rowo = (size_t)b * (size_t)(T_ - 1) * VOC_ + (size_t)t * VOC_;
        int v0 = (nt0 + 0) * 16 + nl;
        if (v0 < VOC_) out[rowo + v0] = acc0[r] + bcls[v0];
        int v1 = (nt0 + 1) * 16 + nl;
        if (has1 && v1 < VOC_) out[rowo + v1] = acc1[r] + bcls[v1];
    }
}

// ---------- launch ----------
extern "C" void kernel_launch(void* const* d_in, const int* in_sizes, int n_in,
                              void* d_out, int out_size, void* d_ws, size_t ws_size,
                              hipStream_t stream) {
    const float* lang = (const float*)d_in[0];   // [256,32,300]
    const float* obj  = (const float*)d_in[1];   // [256,256,128]
    const float* Wmap = (const float*)d_in[2];   // [128,300]
    const float* bmap = (const float*)d_in[3];   // [300]
    const float* Wih  = (const float*)d_in[4];   // [900,300]
    const float* Whh  = (const float*)d_in[5];   // [900,300]
    const float* bih  = (const float*)d_in[6];   // [900]
    const float* bhh  = (const float*)d_in[7];   // [900]
    const float* Wcls = (const float*)d_in[8];   // [300,3433]
    const float* bcls = (const float*)d_in[9];   // [3433]
    const int*   tgt  = (const int*)d_in[10];    // [256]
    float* out = (float*)d_out;

    char*  ws  = (char*)d_ws;
    size_t off = 0;
    auto alloc = [&](size_t bytes) -> void* {
        void* p = ws + off;
        off = (off + bytes + 255) & ~(size_t)255;
        return p;
    };
    const int GATE_FRAGS = 3 * JT * KCH;               // 570
    const int X_FRAGS    = (T_ - 1) * MT * KCH;        // 4960
    const int H_FRAGS    = MT * KCH;                   // 160
    unsigned* WihP  = (unsigned*)alloc((size_t)GATE_FRAGS * 1024);
    unsigned* WhhP  = (unsigned*)alloc((size_t)GATE_FRAGS * 1024);
    unsigned* WclsP = (unsigned*)alloc((size_t)VT * KCH * 1024);
    unsigned* WmapP = (unsigned*)alloc((size_t)JT * KCHM * 1024);
    unsigned* Xfrag = (unsigned*)alloc((size_t)X_FRAGS * 1024);
    float* H0 = (float*)alloc(((size_t)B_ * HS + 64) * 4);
    float* H1 = (float*)alloc(((size_t)B_ * HS + 64) * 4);
    unsigned* Hf0 = (unsigned*)alloc((size_t)H_FRAGS * 1024);
    unsigned* Hf1 = (unsigned*)alloc((size_t)H_FRAGS * 1024);

    pack_gates_kernel<<<(GATE_FRAGS * 256 + 255) / 256, 256, 0, stream>>>(Wih, WihP);
    pack_gates_kernel<<<(GATE_FRAGS * 256 + 255) / 256, 256, 0, stream>>>(Whh, WhhP);
    pack_direct_kernel<<<(VT * KCH * 256 + 255) / 256, 256, 0, stream>>>(
        Wcls, WclsP, EMB_, VOC_, VT, KCH);
    pack_direct_kernel<<<(JT * KCHM * 256 + 255) / 256, 256, 0, stream>>>(
        Wmap, WmapP, FEAT_, EMB_, JT, KCHM);
    pack_x_kernel<<<(X_FRAGS * 256 + 255) / 256, 256, 0, stream>>>(lang, Xfrag);

    h0_kernel<<<dim3(JT, MT), 32, 0, stream>>>(obj, tgt, WmapP, bmap, H0);
    repack_h_kernel<<<dim3(KCH, MT), 32, 0, stream>>>(H0, Hf0);

    float*    Hc = H0;  float*    Hn = H1;
    unsigned* Fc = Hf0; unsigned* Fn = Hf1;
    const int NGRP = (VT + 1) / 2;   // 108 groups of 2 vocab tiles
    for (int t = 0; t < T_ - 1; ++t) {
        gru_step_kernel<<<dim3(KCH, MT), 32, 0, stream>>>(
            Xfrag, t, WihP, WhhP, bih, bhh, Hc, Fc, Hn, Fn);
        logits_kernel<<<dim3(NGRP, MT), 32, 0, stream>>>(Fn, WclsP, bcls, out, t);
        float* tf = Hc; Hc = Hn; Hn = tf;
        unsigned* tu = Fc; Fc = Fn; Fn = tu;
    }
    (void)in_sizes; (void)n_in; (void)out_size; (void)ws_size;
}
